// RFCBidirectionalPropagation_40973988004129
// MI455X (gfx1250) — matrix operand order repaired and checked by reading, not compiled, not run on hardware
//
#include <hip/hip_runtime.h>
#include <math.h>

// ---------------------------------------------------------------------------
// RFC bidirectional propagation (BasicVSR++-like) for MI455X / gfx1250.
// All convs lowered to implicit GEMM with V_WMMA_F32_16X16X32_BF16.
// GEMM: 4-wave workgroups, B fragments double-buffered in LDS via
// GLOBAL_LOAD_ASYNC_TO_LDS_B64 (ASYNCcnt), 2 M-tiles per wave -> 8x B reuse.
// B=1, T=12, C=128, H=W=64, G=16, K=3.
// ---------------------------------------------------------------------------

#define HW      4096
#define WIMG    64
#define NTILES  256      // HW / 16 pixel tiles

typedef __attribute__((ext_vector_type(16))) __bf16 v16bf;
typedef __attribute__((ext_vector_type(8)))  float  v8f;
typedef int a_v2i __attribute__((vector_size(8)));   // async-copy payload type

__device__ __forceinline__ unsigned short f2bf(float f) {
  union { float f; unsigned int u; } v; v.f = f;
  unsigned int u = v.u;
  if ((u & 0x7F800000u) == 0x7F800000u) return (unsigned short)(u >> 16); // inf/nan
  return (unsigned short)((u + 0x7FFFu + ((u >> 16) & 1u)) >> 16);        // RNE
}

// Async global->LDS staging (8 bytes per thread). Uses the CDNA5 async-copy
// path (no VGPR transit, tracked by ASYNCcnt) when the builtin is available.
__device__ __forceinline__ void stage_b64(const unsigned short* g, unsigned short* l) {
#if __has_builtin(__builtin_amdgcn_global_load_async_to_lds_b64)
  __builtin_amdgcn_global_load_async_to_lds_b64(
      (__attribute__((address_space(1))) a_v2i*)g,
      (__attribute__((address_space(3))) a_v2i*)l, 0, 0);
#else
  *(uint2*)l = *(const uint2*)g;
#endif
}
__device__ __forceinline__ void stage_wait() {
#if __has_builtin(__builtin_amdgcn_s_wait_asynccnt)
  __builtin_amdgcn_s_wait_asynccnt(0);
#endif
}

// ---------------------------------------------------------------------------
// Pack conv weights w[Cout][K] (K = Cin*9 flat OIHW, or Cin for 1x1) into
// WMMA A-fragment tiles: out[mt][kt][lane][j], 16 bf16 per lane.
// M rows are zero-padded up to Mtiles_pad*16 (Cout passed for the guard).
// A 16x32 bf16 layout (matches B packing): lane<16: j0..15 -> K 0..15
//                                   lane>=16: j0..15 -> K 16..31
// ---------------------------------------------------------------------------
__global__ void pack_w_kernel(const float* __restrict__ w, unsigned short* __restrict__ out,
                              int Ktiles, int K, int Cout, long total) {
  long idx = (long)blockIdx.x * blockDim.x + threadIdx.x;
  if (idx >= total) return;
  long tile = idx >> 9; int e = (int)(idx & 511);
  int lane = e >> 4, j = e & 15;
  int kt = (int)(tile % Ktiles);
  int mt = (int)(tile / Ktiles);
  int hi = (lane >= 16) ? 16 : 0;
  int kk = j + hi;
  int Kidx = kt * 32 + kk;
  int M = mt * 16 + (lane & 15);
  float v = (M < Cout) ? w[(long)M * K + Kidx] : 0.f;
  out[idx] = f2bf(v);
}

// ---------------------------------------------------------------------------
// im2col for 3x3 pad-1 (khw==9) or 1x1 (khw==1) convs over up to 3
// channel-concatenated sources (null source == zeros). Output is the WMMA
// B-fragment layout: out[kt][nt][lane][j]; lane<16: col=lane, K=kt*32+j;
// lane>=16: col=lane-16, K=kt*32+16+j.
// ---------------------------------------------------------------------------
__global__ void im2col_kernel(const float* s0, int c0, const float* s1, int c1,
                              const float* s2, int c2, unsigned short* __restrict__ out,
                              int Ktiles, int khw, long total) {
  long idx = (long)blockIdx.x * blockDim.x + threadIdx.x;
  if (idx >= total) return;
  long tile = idx >> 9; int e = (int)(idx & 511);
  int lane = e >> 4, j = e & 15;
  int nt = (int)(tile % NTILES);
  int kt = (int)(tile / NTILES);
  int K = kt * 32 + ((lane >> 4) << 4) + j;
  int n = nt * 16 + (lane & 15);
  int y = n >> 6, x = n & 63;
  int cin, ky, kx;
  if (khw == 9) { cin = K / 9; int r = K - cin * 9; int r3 = r / 3; ky = r3 - 1; kx = r - r3 * 3 - 1; }
  else          { cin = K; ky = 0; kx = 0; }
  const float* src; int c;
  if (cin < c0)           { src = s0; c = cin; }
  else if (cin < c0 + c1) { src = s1; c = cin - c0; }
  else                    { src = s2; c = cin - c0 - c1; }
  int yy = y + ky, xx = x + kx;
  float v = 0.f;
  if (src && (unsigned)yy < 64u && (unsigned)xx < 64u)
    v = src[(long)c * HW + yy * WIMG + xx];
  out[idx] = f2bf(v);
}

// ---------------------------------------------------------------------------
// GEMM: out[Cout][4096] = A[Cout_pad][K] x B[K][4096] + bias (+optional
// residual, optional leaky ReLU).
// Block = 128 threads (4 waves). All waves share one pixel tile nt; the
// 1KB B fragment for each kt is staged in LDS (double-buffered, async copy)
// by the whole block, then each wave runs 2 WMMAs against 2 A fragments
// (2 M-tiles/wave, 8 M-tiles/block).
// C/D layout: VGPR r: lanes0-15 M=r, lanes16-31 M=8+r; N=lane&15.
// ---------------------------------------------------------------------------
__global__ void gemm_wmma_kernel(const unsigned short* __restrict__ A,
                                 const unsigned short* __restrict__ B,
                                 const float* __restrict__ bias,
                                 float* __restrict__ out, int Ktiles, int act,
                                 const float* __restrict__ addsrc, int Cout) {
  __shared__ __align__(32) unsigned short sB[2][512];
  int nt   = blockIdx.x;        // pixel tile 0..255
  int tid  = threadIdx.x;       // 0..127
  int wave = tid >> 5;
  int lane = tid & 31;
  int mt0  = blockIdx.y * 8 + wave * 2;

  const v16bf* Ap  = (const v16bf*)A + ((long)mt0 * Ktiles) * 32 + lane;
  const long   Amt = (long)Ktiles * 32;   // v16bf stride between M-tiles

  // stage kt = 0 (each thread async-copies 8 bytes of the 1KB fragment)
  stage_b64(B + (long)nt * 512 + tid * 4, &sB[0][tid * 4]);
  stage_wait();
  __syncthreads();

  v8f acc0 = {}, acc1 = {};
  for (int kt = 0; kt < Ktiles; ++kt) {
    int cur = kt & 1;
    if (kt + 1 < Ktiles) {   // async-prefetch next B fragment into other buffer
      stage_b64(B + ((long)(kt + 1) * NTILES + nt) * 512 + tid * 4,
                &sB[cur ^ 1][tid * 4]);
    }
    v16bf b  = *(const v16bf*)&sB[cur][lane * 16];
    v16bf a0 = Ap[(long)kt * 32];
    v16bf a1 = Ap[(long)kt * 32 + Amt];
    acc0 = __builtin_amdgcn_wmma_f32_16x16x32_bf16(false, a0, false, b,
                                                   (short)0, acc0, false, false);
    acc1 = __builtin_amdgcn_wmma_f32_16x16x32_bf16(false, a1, false, b,
                                                   (short)0, acc1, false, false);
    stage_wait();            // drain this wave's async copy before the barrier
    __syncthreads();
  }

  int px    = nt * 16 + (lane & 15);
  int rbase = (lane >> 4) << 3;
#pragma unroll
  for (int r = 0; r < 8; ++r) {
    int ch = mt0 * 16 + rbase + r;
    if (ch < Cout) {
      float v = acc0[r] + bias[ch];
      if (act) v = (v >= 0.f) ? v : 0.1f * v;
      long o = (long)ch * HW + px;
      if (addsrc) v += addsrc[o];
      out[o] = v;
    }
  }
#pragma unroll
  for (int r = 0; r < 8; ++r) {
    int ch = (mt0 + 1) * 16 + rbase + r;
    if (ch < Cout) {
      float v = acc1[r] + bias[ch];
      if (act) v = (v >= 0.f) ? v : 0.1f * v;
      long o = (long)ch * HW + px;
      if (addsrc) v += addsrc[o];
      out[o] = v;
    }
  }
}

// ---------------------------------------------------------------------------
// offset/mask activation: y has 432 channels; ch<288 -> 5*tanh (offsets),
// ch>=288 -> sigmoid (masks).
// ---------------------------------------------------------------------------
__global__ void offact_kernel(const float* __restrict__ y, float* __restrict__ off,
                              float* __restrict__ mask, long total) {
  long idx = (long)blockIdx.x * blockDim.x + threadIdx.x;
  if (idx >= total) return;
  float v = y[idx];
  if (idx < (long)288 * HW) off[idx] = 5.0f * tanhf(v);
  else mask[idx - (long)288 * HW] = 1.0f / (1.0f + expf(-v));
}

// ---------------------------------------------------------------------------
// Deformable sampling: builds the masked bilinear-sampled "im2col" matrix
// S[2304][4096] (G=16, Cg=16, 3x3) directly in WMMA B-fragment layout.
// x_in = concat(p1, p2); null p2 == zeros. offset ch = (g*9+k)*2 + {dy,dx},
// mask ch = g*9+k. OOB corners contribute zero (matches reference valid-mask).
// ---------------------------------------------------------------------------
__global__ void dsample_kernel(const float* p1, const float* p2,
                               const float* __restrict__ off,
                               const float* __restrict__ mask,
                               unsigned short* __restrict__ out, long total) {
  long idx = (long)blockIdx.x * blockDim.x + threadIdx.x;
  if (idx >= total) return;
  long tile = idx >> 9; int e = (int)(idx & 511);
  int lane = e >> 4, j = e & 15;
  int nt = (int)(tile % NTILES);
  int kt = (int)(tile / NTILES);
  int K = kt * 32 + ((lane >> 4) << 4) + j;     // 0..2303
  int n = nt * 16 + (lane & 15);
  int y = n >> 6, x = n & 63;
  int cin = K / 9; int r = K - cin * 9;         // cin 0..255
  int g = cin >> 4;                             // Cg = 16
  int r3 = r / 3;
  int ky = r3 - 1, kx = r - r3 * 3 - 1;
  int oc = g * 9 + r;
  float dy = off[(long)(oc * 2 + 0) * HW + n];
  float dx = off[(long)(oc * 2 + 1) * HW + n];
  float m  = mask[(long)oc * HW + n];
  const float* src = (cin < 128) ? p1 : p2;
  float val = 0.f;
  if (src) {
    const float* sc = src + (long)((cin < 128) ? cin : cin - 128) * HW;
    float py = (float)(y + ky) + dy;
    float px = (float)(x + kx) + dx;
    float fy = floorf(py), fx = floorf(px);
    float wy = py - fy, wx = px - fx;
    int y0 = (int)fy, x0 = (int)fx;
    float a00 = 0.f, a01 = 0.f, a10 = 0.f, a11 = 0.f;
    if ((unsigned)y0 < 64u) {
      if ((unsigned)x0 < 64u)       a00 = sc[y0 * 64 + x0];
      if ((unsigned)(x0 + 1) < 64u) a01 = sc[y0 * 64 + x0 + 1];
    }
    if ((unsigned)(y0 + 1) < 64u) {
      if ((unsigned)x0 < 64u)       a10 = sc[(y0 + 1) * 64 + x0];
      if ((unsigned)(x0 + 1) < 64u) a11 = sc[(y0 + 1) * 64 + x0 + 1];
    }
    val = (1.f - wy) * ((1.f - wx) * a00 + wx * a01)
        +        wy  * ((1.f - wx) * a10 + wx * a11);
  }
  out[idx] = f2bf(val * m);
}

// ---------------------------------------------------------------------------
// Host orchestration
// ---------------------------------------------------------------------------
extern "C" void kernel_launch(void* const* d_in, const int* in_sizes, int n_in,
                              void* d_out, int out_size, void* d_ws, size_t ws_size,
                              hipStream_t stream) {
  const int  C   = 128;
  const int  T   = 12;
  const long CHW = (long)C * HW;   // 524288

  const float* X   = (const float*)d_in[0];
  float*       OUT = (float*)d_out;
  auto P = [&](int i) { return (const float*)d_in[i]; };

  char* ws = (char*)d_ws;
  size_t pos = 0;
  auto alloc = [&](size_t bytes) -> char* {
    char* p = ws + pos; pos += (bytes + 255) & ~(size_t)255; return p;
  };

  struct PW { unsigned short* p; int Cout; int K; int MtPad; };
  auto packw = [&](const float* w, int Cout, int K) -> PW {
    PW r; r.Cout = Cout; r.K = K;
    int Mt = (Cout + 15) / 16;
    r.MtPad = (Mt + 7) & ~7;                        // multiple of 8 M-tiles
    r.p = (unsigned short*)alloc((size_t)r.MtPad * 16 * K * sizeof(unsigned short));
    long total = (long)r.MtPad * (K / 32) * 512;
    pack_w_kernel<<<dim3((unsigned)((total + 255) / 256)), 256, 0, stream>>>(
        w, r.p, K / 32, K, Cout, total);
    return r;
  };

  const int bB = 1, bF = 15;   // param base indices (bwd_, fwd_)
  PW b_off1 = packw(P(bB + 0),  C,   3 * C * 9);
  PW b_off2 = packw(P(bB + 2),  C,   C * 9);
  PW b_off3 = packw(P(bB + 4),  C,   C * 9);
  PW b_off4 = packw(P(bB + 6),  432, C * 9);
  PW b_dc   = packw(P(bB + 8),  C,   2 * C * 9);
  PW b_bb1  = packw(P(bB + 10), C,   2 * C * 9);
  PW b_bb2  = packw(P(bB + 12), C,   C * 9);
  PW f_off1 = packw(P(bF + 0),  C,   3 * C * 9);
  PW f_off2 = packw(P(bF + 2),  C,   C * 9);
  PW f_off3 = packw(P(bF + 4),  C,   C * 9);
  PW f_off4 = packw(P(bF + 6),  432, C * 9);
  PW f_dc   = packw(P(bF + 8),  C,   2 * C * 9);
  PW f_bb1  = packw(P(bF + 10), C,   3 * C * 9);  // fwd resblock has 3C input
  PW f_bb2  = packw(P(bF + 12), C,   C * 9);
  PW fusw   = packw(P(29),      C,   2 * C);      // 1x1 fusion

  float* bwd[12]; float* fwd[12];
  for (int t = 0; t < T; ++t) bwd[t] = (float*)alloc(CHW * 4);
  for (int t = 0; t < T; ++t) fwd[t] = (float*)alloc(CHW * 4);
  float* h1    = (float*)alloc(CHW * 4);
  float* h2    = (float*)alloc(CHW * 4);
  float* h3    = (float*)alloc(CHW * 4);
  float* y4    = (float*)alloc((size_t)432 * HW * 4);
  float* offb  = (float*)alloc((size_t)288 * HW * 4);
  float* maskb = (float*)alloc((size_t)144 * HW * 4);
  float* fpdc  = (float*)alloc(CHW * 4);
  float* rh    = (float*)alloc(CHW * 4);
  unsigned short* Scol = (unsigned short*)alloc((size_t)108 * NTILES * 512 * 2); // max Ktiles=108

  auto gemm = [&](const PW& Aw, const float* bias, float* out, int act,
                  const float* addsrc) {
    gemm_wmma_kernel<<<dim3(NTILES, Aw.MtPad / 8), 128, 0, stream>>>(
        Aw.p, Scol, bias, out, Aw.K / 32, act, addsrc, Aw.Cout);
  };

  auto conv = [&](const float* s0, int c0, const float* s1, int c1,
                  const float* s2, int c2, const PW& Aw, const float* bias,
                  float* out, int act, const float* addsrc, int khw) {
    int Kt = Aw.K / 32;
    long total = (long)Kt * NTILES * 512;
    im2col_kernel<<<dim3((unsigned)((total + 255) / 256)), 256, 0, stream>>>(
        s0, c0, s1, c1, s2, c2, Scol, Kt, khw, total);
    gemm(Aw, bias, out, act, addsrc);
  };

  const float* xt[12];
  for (int t = 0; t < T; ++t) xt[t] = X + (long)t * CHW;
  const float* NUL = nullptr;

  // ---------------- backward propagation (reversed time) ----------------
  // fp0 = resblock(concat(x[11], zeros))
  conv(xt[11], C, NUL, C, NUL, 0, b_bb1, P(bB + 11), rh, 1, nullptr, 9);
  conv(rh, C, NUL, 0, NUL, 0, b_bb2, P(bB + 13), bwd[11], 0, nullptr, 9);
  for (int i = 1; i < T; ++i) {
    const float* cur = xt[11 - i];
    const float* p1  = bwd[12 - i];
    const float* p2  = (i >= 2) ? bwd[13 - i] : nullptr;
    // conv_offset on cond = concat(p1, cur, p2)
    conv(p1, C, cur, C, p2, C, b_off1, P(bB + 1), h1, 1, nullptr, 9);
    conv(h1, C, NUL, 0, NUL, 0, b_off2, P(bB + 3), h2, 1, nullptr, 9);
    conv(h2, C, NUL, 0, NUL, 0, b_off3, P(bB + 5), h3, 1, nullptr, 9);
    conv(h3, C, NUL, 0, NUL, 0, b_off4, P(bB + 7), y4, 0, nullptr, 9);
    { long tot = (long)432 * HW;
      offact_kernel<<<(unsigned)((tot + 255) / 256), 256, 0, stream>>>(y4, offb, maskb, tot); }
    { long tot = (long)72 * NTILES * 512;
      dsample_kernel<<<(unsigned)((tot + 255) / 256), 256, 0, stream>>>(p1, p2, offb, maskb, Scol, tot); }
    gemm(b_dc, P(bB + 9), fpdc, 0, nullptr);
    // fp = fpdc + resblock(concat(cur, fpdc))
    conv(cur, C, fpdc, C, NUL, 0, b_bb1, P(bB + 11), rh, 1, nullptr, 9);
    conv(rh, C, NUL, 0, NUL, 0, b_bb2, P(bB + 13), bwd[11 - i], 0, fpdc, 9);
  }

  // ---------------- forward propagation (extra = bwd) ----------------
  conv(xt[0], C, bwd[0], C, NUL, C, f_bb1, P(bF + 11), rh, 1, nullptr, 9);
  conv(rh, C, NUL, 0, NUL, 0, f_bb2, P(bF + 13), fwd[0], 0, nullptr, 9);
  for (int t = 1; t < T; ++t) {
    const float* cur = xt[t];
    const float* ext = bwd[t];
    const float* p1  = fwd[t - 1];
    const float* p2  = (t >= 2) ? fwd[t - 2] : nullptr;
    conv(p1, C, cur, C, p2, C, f_off1, P(bF + 1), h1, 1, nullptr, 9);
    conv(h1, C, NUL, 0, NUL, 0, f_off2, P(bF + 3), h2, 1, nullptr, 9);
    conv(h2, C, NUL, 0, NUL, 0, f_off3, P(bF + 5), h3, 1, nullptr, 9);
    conv(h3, C, NUL, 0, NUL, 0, f_off4, P(bF + 7), y4, 0, nullptr, 9);
    { long tot = (long)432 * HW;
      offact_kernel<<<(unsigned)((tot + 255) / 256), 256, 0, stream>>>(y4, offb, maskb, tot); }
    { long tot = (long)72 * NTILES * 512;
      dsample_kernel<<<(unsigned)((tot + 255) / 256), 256, 0, stream>>>(p1, p2, offb, maskb, Scol, tot); }
    gemm(f_dc, P(bF + 9), fpdc, 0, nullptr);
    conv(cur, C, ext, C, fpdc, C, f_bb1, P(bF + 11), rh, 1, nullptr, 9);
    conv(rh, C, NUL, 0, NUL, 0, f_bb2, P(bF + 13), fwd[t], 0, fpdc, 9);
  }

  // ---------------- fusion: 1x1 conv(concat(bwd, fwd)) + x ----------------
  for (int t = 0; t < T; ++t) {
    conv(bwd[t], C, fwd[t], C, NUL, 0, fusw, P(30), OUT + (long)t * CHW, 0, xt[t], 1);
  }
}
